// kernel_layer_52140902973634
// MI455X (gfx1250) — matrix operand order repaired and checked by reading, not compiled, not run on hardware
//
#include <hip/hip_runtime.h>
#include <hip/hip_bf16.h>
#include <cstddef>

// Problem constants (from reference)
#define BB 64
#define LL 512
#define DD 768
#define BD (BB * DD)   // 49152
#define BL (BB * LL)   // 32768

typedef __attribute__((ext_vector_type(16))) _Float16 v16h;
typedef __attribute__((ext_vector_type(8)))  float    v8f;

// ---------------------------------------------------------------- zero scratch
__global__ __launch_bounds__(256) void k0_zero(float* ws, int n) {
    int i = blockIdx.x * 256 + threadIdx.x;
    if (i < n) ws[i] = 0.f;
}

// ---------------------------------------------------------------- pass 1:
// cos[b,l] + fused per-(b,d) column sums (all l) and edge sums (l==0, l==511).
// One wave per row; 8 rows (one b) per 256-thread block.
__global__ __launch_bounds__(256) void k1_cos_colsum(
        const float* __restrict__ news, const float* __restrict__ post,
        float* __restrict__ cosv,
        float* __restrict__ colsum_n, float* __restrict__ colsum_p,
        float* __restrict__ edge_n,   float* __restrict__ edge_p) {
    __shared__ float sn[DD], sp[DD];
    const int b    = blockIdx.y;
    const int w    = threadIdx.x >> 5;
    const int lane = threadIdx.x & 31;
    const int l    = blockIdx.x * 8 + w;

    for (int i = threadIdx.x; i < DD; i += 256) { sn[i] = 0.f; sp[i] = 0.f; }
    __syncthreads();

    const float* nrow = news + ((size_t)b * LL + l) * DD;
    const float* prow = post + ((size_t)b * LL + l) * DD;
    const bool edge = (l == 0) || (l == LL - 1);

    float dot = 0.f, n2 = 0.f, p2 = 0.f;
    for (int d = lane; d < DD; d += 32) {
        float nv = nrow[d], pv = prow[d];
        dot += nv * pv; n2 += nv * nv; p2 += pv * pv;
        atomicAdd(&sn[d], nv);                       // ds_add_f32
        atomicAdd(&sp[d], pv);
        if (edge) {
            unsafeAtomicAdd(&edge_n[b * DD + d], nv); // global_atomic_add_f32
            unsafeAtomicAdd(&edge_p[b * DD + d], pv);
        }
    }
    // wave32 reduction
    for (int off = 16; off; off >>= 1) {
        dot += __shfl_xor(dot, off, 32);
        n2  += __shfl_xor(n2,  off, 32);
        p2  += __shfl_xor(p2,  off, 32);
    }
    if (lane == 0) {
        float nn = fmaxf(sqrtf(n2), 1e-8f);
        float pn = fmaxf(sqrtf(p2), 1e-8f);
        cosv[b * LL + l] = dot / (nn * pn);
    }
    __syncthreads();
    for (int i = threadIdx.x; i < DD; i += 256) {
        unsafeAtomicAdd(&colsum_n[b * DD + i], sn[i]);
        unsafeAtomicAdd(&colsum_p[b * DD + i], sp[i]);
    }
}

// ---------------------------------------------------------------- pass 2:
// 1-D conv of cos with column-summed 3x3 Gaussian -> wk_int / wk_edge rows,
// plus global BN sum / sumsq (weights 510 and 2).
__global__ __launch_bounds__(512) void k2_conv_bnstats(
        const float* __restrict__ cosv, const float* __restrict__ g,
        float* __restrict__ wk2, float* __restrict__ bn_stats) {
    __shared__ float red0[16], red1[16];
    const int b = blockIdx.x;
    const int j = threadIdx.x;

    float gc0 = g[0] + g[3] + g[6];   // full column sums (interior rows)
    float gc1 = g[1] + g[4] + g[7];
    float gc2 = g[2] + g[5] + g[8];
    float ge0 = g[3] + g[6];          // edge row (top==bottom, Gaussian symm.)
    float ge1 = g[4] + g[7];
    float ge2 = g[5] + g[8];

    float cm = (j > 0)      ? cosv[b * LL + j - 1] : 0.f;
    float cc =                cosv[b * LL + j];
    float cp = (j < LL - 1) ? cosv[b * LL + j + 1] : 0.f;

    float wi = gc0 * cm + gc1 * cc + gc2 * cp;
    float we = ge0 * cm + ge1 * cc + ge2 * cp;
    wk2[(2 * b) * LL + j]     = wi;
    wk2[(2 * b + 1) * LL + j] = we;

    float s  = 510.f * wi + 2.f * we;
    float sq = 510.f * wi * wi + 2.f * we * we;
    for (int off = 16; off; off >>= 1) {
        s  += __shfl_xor(s,  off, 32);
        sq += __shfl_xor(sq, off, 32);
    }
    int wv = j >> 5, ln = j & 31;
    if (ln == 0) { red0[wv] = s; red1[wv] = sq; }
    __syncthreads();
    if (wv == 0) {
        float ts = (ln < 16) ? red0[ln] : 0.f;
        float tq = (ln < 16) ? red1[ln] : 0.f;
        for (int off = 16; off; off >>= 1) {
            ts += __shfl_xor(ts, off, 32);
            tq += __shfl_xor(tq, off, 32);
        }
        if (ln == 0) {
            unsafeAtomicAdd(&bn_stats[0], ts);
            unsafeAtomicAdd(&bn_stats[1], tq);
        }
    }
}

// ---------------------------------------------------------------- pass 3:
// y[r,n] = sum_m ( a*wk2[r,m]+c ) * W[n,m] + lb[n],  r = 0..127 (2 rows/batch)
// WMMA f32<-f16 16x16x32; one wave per 16x16 output tile; K=512 -> 16 WMMAs.
__global__ __launch_bounds__(32) void k3_wmma_gemm(
        const float* __restrict__ wk2,      // [128][512]
        const float* __restrict__ lin_w,    // [512][512], (n, m)
        const float* __restrict__ lin_b,    // [512]
        const float* __restrict__ bn_stats,
        const float* __restrict__ bn_gamma, const float* __restrict__ bn_beta,
        float* __restrict__ y_all) {        // [128][512]
    const int tile  = blockIdx.x;      // 0..255
    const int mtile = tile & 7;        // 8 tiles over 128 rows
    const int ntile = tile >> 3;       // 32 tiles over 512 cols
    const int lane  = threadIdx.x;     // wave32, EXEC all ones
    const int half  = lane >> 4;
    const int l16   = lane & 15;

    const float Ntot = (float)BB * (float)LL * (float)LL;
    float mu  = bn_stats[0] / Ntot;
    float var = bn_stats[1] / Ntot - mu * mu;
    float a   = bn_gamma[0] * rsqrtf(var + 1e-5f);
    float c0  = bn_beta[0] - a * mu;

    const int arow = mtile * 16 + l16;   // A row owned by this lane (ISA 7.12.2)
    const int nb   = ntile * 16 + l16;   // B column (== W row) for this lane

    v8f acc = {};
    for (int k0 = 0; k0 < LL; k0 += 32) {
        v16h af, bf;
#pragma unroll
        for (int e = 0; e < 16; ++e) {
            // A 16x32 f16 layout: e<8 -> K = 8*half+e ; e>=8 -> K = 16+8*half+(e-8)
            int kA = (e < 8) ? (8 * half + e) : (16 + 8 * half + (e - 8));
            af[e] = (_Float16)(a * wk2[arow * LL + k0 + kA] + c0);
            // B 32x16 f16 layout: same K order as A, N = lane%16
            int kB = (e < 8) ? (8 * half + e) : (16 + 8 * half + (e - 8));
            bf[e] = (_Float16)lin_w[nb * LL + k0 + kB];
        }
        acc = __builtin_amdgcn_wmma_f32_16x16x32_f16(
                  false, af, false, bf, (short)0, acc, false, false);
    }
    // D layout: VGPR v, lanes 0-15 -> M=v ; lanes 16-31 -> M=v+8 ; N = lane%16
#pragma unroll
    for (int v = 0; v < 8; ++v) {
        int r = mtile * 16 + v + 8 * half;
        int n = ntile * 16 + l16;
        y_all[r * LL + n] = acc[v] + lin_b[n];
    }
}

// ---------------------------------------------------------------- pass 4:
// 2-point softmax over l (multiplicities 510 / 2), summed over n per batch.
__global__ __launch_bounds__(512) void k4_softmax_sum(
        const float* __restrict__ y_all, float* __restrict__ s_int,
        float* __restrict__ s_edge) {
    __shared__ float red0[16], red1[16];
    const int b = blockIdx.x, t = threadIdx.x;
    float yi = y_all[(2 * b) * LL + t];
    float ye = y_all[(2 * b + 1) * LL + t];
    float m  = fmaxf(yi, ye);
    float ei = __expf(yi - m), ee = __expf(ye - m);
    float denom = 2.f * ee + 510.f * ei;
    float si = ei / denom, se = ee / denom;
    for (int off = 16; off; off >>= 1) {
        si += __shfl_xor(si, off, 32);
        se += __shfl_xor(se, off, 32);
    }
    int wv = t >> 5, ln = t & 31;
    if (ln == 0) { red0[wv] = si; red1[wv] = se; }
    __syncthreads();
    if (wv == 0) {
        float ts = (ln < 16) ? red0[ln] : 0.f;
        float te = (ln < 16) ? red1[ln] : 0.f;
        for (int off = 16; off; off >>= 1) {
            ts += __shfl_xor(ts, off, 32);
            te += __shfl_xor(te, off, 32);
        }
        if (ln == 0) { s_int[b] = ts; s_edge[b] = te; }
    }
}

// ---------------------------------------------------------------- pass 5:
// word_feat[b,d] = (se*E + si*(S-E))/512 with column/edge sums from pass 1.
__global__ __launch_bounds__(256) void k5_final(
        const float* __restrict__ colsum_n, const float* __restrict__ colsum_p,
        const float* __restrict__ edge_n,   const float* __restrict__ edge_p,
        const float* __restrict__ s_int,    const float* __restrict__ s_edge,
        float* __restrict__ out) {
    int idx = blockIdx.x * 256 + threadIdx.x;
    if (idx >= BD) return;
    int b = idx / DD;
    float si = s_int[b], se = s_edge[b];
    float en = edge_n[idx], ep = edge_p[idx];
    out[idx]      = (se * en + si * (colsum_n[idx] - en)) * (1.f / (float)LL);
    out[BD + idx] = (se * ep + si * (colsum_p[idx] - ep)) * (1.f / (float)LL);
}

// ---------------------------------------------------------------- launch
extern "C" void kernel_launch(void* const* d_in, const int* in_sizes, int n_in,
                              void* d_out, int out_size, void* d_ws, size_t ws_size,
                              hipStream_t stream) {
    const float* news  = (const float*)d_in[0];
    const float* post  = (const float*)d_in[1];
    const float* gk    = (const float*)d_in[2];
    const float* gamma = (const float*)d_in[3];
    const float* beta  = (const float*)d_in[4];
    const float* lw    = (const float*)d_in[5];
    const float* lb    = (const float*)d_in[6];
    float* out = (float*)d_out;

    float* ws       = (float*)d_ws;
    float* colsum_n = ws;                 // BD
    float* colsum_p = colsum_n + BD;      // BD
    float* edge_n   = colsum_p + BD;      // BD
    float* edge_p   = edge_n + BD;        // BD
    float* bn_stats = edge_p + BD;        // 16 (2 used)
    float* cosv     = bn_stats + 16;      // BL
    float* wk2      = cosv + BL;          // 2*BL
    float* y_all    = wk2 + 2 * BL;       // 2*BL
    float* s_int    = y_all + 2 * BL;     // BB
    float* s_edge   = s_int + BB;         // BB
    // total ~1.44 MB of scratch

    const int ZN = 4 * BD + 16;           // atomically-accumulated region
    k0_zero<<<(ZN + 255) / 256, 256, 0, stream>>>(ws, ZN);

    k1_cos_colsum<<<dim3(LL / 8, BB), 256, 0, stream>>>(
        news, post, cosv, colsum_n, colsum_p, edge_n, edge_p);

    k2_conv_bnstats<<<BB, 512, 0, stream>>>(cosv, gk, wk2, bn_stats);

    k3_wmma_gemm<<<256, 32, 0, stream>>>(wk2, lw, lb, bn_stats, gamma, beta, y_all);

    k4_softmax_sum<<<BB, 512, 0, stream>>>(y_all, s_int, s_edge);

    k5_final<<<(BD + 255) / 256, 256, 0, stream>>>(
        colsum_n, colsum_p, edge_n, edge_p, s_int, s_edge, out);
}
